// MyBasicNei_3702261809840
// MI455X (gfx1250) — matrix operand order repaired and checked by reading, not compiled, not run on hardware
//
#include <hip/hip_runtime.h>
#include <hip/hip_bf16.h>

// MI455X plan: ~44 GFLOP dominated by per-neighbor 2-layer MLPs on 262144 gathered
// rows (x2 blocks). ~100MB of traffic -> ~4.3us at 23.3TB/s; bf16 WMMA keeps the
// GEMM side in the same ballpark, so everything runs as fused bf16 WMMA tiles with
// f32 accumulation and no big intermediates in HBM. This round: native bf16 cvt
// (v_cvt_pk_bf16_f32 path), A-fragment reuse across k/v passes, vectorized LDS
// reads of softmax weights.

typedef __attribute__((ext_vector_type(16))) __bf16 v16bf;
typedef __attribute__((ext_vector_type(2)))  __bf16 v2bf;
typedef __attribute__((ext_vector_type(8)))  float  v8f;

union Frag { v16bf v; unsigned int u32[8]; };

#define DEV static __device__ __forceinline__

DEV unsigned short f2bf(float a){
  __bf16 t = (__bf16)a;                       // RNE, native cvt on gfx1250
  unsigned short u; __builtin_memcpy(&u, &t, 2); return u;
}
DEV unsigned int pack2(float a, float b){
  v2bf t; t.x = (__bf16)a; t.y = (__bf16)b;   // -> v_cvt_pk_bf16_f32
  unsigned int u; __builtin_memcpy(&u, &t, 4); return u;
}
DEV float eluf(float x){ return x > 0.f ? x : (__expf(x) - 1.f); }

DEV v8f wmma_bf(const Frag& A, const Frag& B, v8f C){
  return __builtin_amdgcn_wmma_f32_16x16x32_bf16(false, A.v, false, B.v,
                                                 (short)0, C, false, false);
}

// A-matrix 16x32 bf16 layout (ISA 7.12.2): lane m=L&15; VGPR j packs K pair
// kbase(j,half) = (j<4 ? 2j : 16+2(j-4)) + 8*half
DEV int kpatA(int j, int half){
  return ((j < 4) ? (j << 1) : (16 + ((j - 4) << 1))) + (half << 3);
}

DEV void load_afrag_f32(const float* src, int half, Frag& A){
#pragma unroll
  for (int j = 0; j < 8; ++j){
    int kb = kpatA(j, half);
    float2 f = *(const float2*)(src + kb);
    A.u32[j] = pack2(f.x, f.y);
  }
}
DEV void load_afrag_bf16(const unsigned short* src, int half, Frag& A){
#pragma unroll
  for (int j = 0; j < 8; ++j){
    int kb = kpatA(j, half);
    A.u32[j] = *(const unsigned int*)(src + kb);
  }
}
// B fragments are pre-permuted so each lane's 16 elements are contiguous (32B).
DEV void load_bfrag(const unsigned short* frag, int kc, int nt, int lane, Frag& B){
  const uint4* p = (const uint4*)(frag + ((size_t)((kc * 8 + nt) * 32 + lane) << 4));
  uint4 a = p[0], b = p[1];
  B.u32[0]=a.x; B.u32[1]=a.y; B.u32[2]=a.z; B.u32[3]=a.w;
  B.u32[4]=b.x; B.u32[5]=b.y; B.u32[6]=b.z; B.u32[7]=b.w;
}

// Store D-layout accumulators (post bias+elu) as bf16 row-major [16][128] in LDS.
DEV void hid_store(const v8f* acc, const float* b1, unsigned short* hidw,
                   int l15, int half){
#pragma unroll
  for (int nt = 0; nt < 8; ++nt){
    float bv = b1[nt * 16 + l15];
#pragma unroll
    for (int v = 0; v < 8; v += 2){
      unsigned int p = pack2(eluf(acc[nt][v] + bv), eluf(acc[nt][v + 1] + bv));
      hidw[(v     + 8 * half) * 128 + nt * 16 + l15] = (unsigned short)p;
      hidw[(v + 1 + 8 * half) * 128 + nt * 16 + l15] = (unsigned short)(p >> 16);
    }
  }
  asm volatile("s_wait_dscnt 0" ::: "memory");   // stores visible before A reads
}

DEV void mlp_l1_fragA(const Frag& A, const unsigned short* W1f, int lane, v8f* acc){
  v8f z = {};
#pragma unroll
  for (int nt = 0; nt < 8; ++nt){
    Frag B; load_bfrag(W1f, 0, nt, lane, B);
    acc[nt] = wmma_bf(A, B, z);
  }
}
DEV void mlp_l1_bf16(const unsigned short* arow, const unsigned short* W1f,
                     int lane, int half, v8f* acc){
  v8f z = {};
#pragma unroll
  for (int nt = 0; nt < 8; ++nt) acc[nt] = z;
#pragma unroll
  for (int kc = 0; kc < 4; ++kc){
    Frag A; load_afrag_bf16(arow + kc * 32, half, A);
#pragma unroll
    for (int nt = 0; nt < 8; ++nt){
      Frag B; load_bfrag(W1f, kc, nt, lane, B);
      acc[nt] = wmma_bf(A, B, acc[nt]);
    }
  }
}
DEV void mlp_l2(const unsigned short* W2f, const unsigned short* hidw,
                int lane, int l15, int half, v8f* acc2){
  v8f z = {};
#pragma unroll
  for (int nt = 0; nt < 8; ++nt) acc2[nt] = z;
#pragma unroll
  for (int kc = 0; kc < 4; ++kc){
    Frag A; load_afrag_bf16(hidw + l15 * 128 + kc * 32, half, A);
#pragma unroll
    for (int nt = 0; nt < 8; ++nt){
      Frag B; load_bfrag(W2f, kc, nt, lane, B);
      acc2[nt] = wmma_bf(A, B, acc2[nt]);
    }
  }
}

// ---- weight permutation: f32 [K,128] -> fragment-ordered bf16 -------------
__global__ void k_prep(const float* __restrict__ W, unsigned short* __restrict__ out,
                       int total){
  int t = blockIdx.x * 256 + threadIdx.x;
  if (t >= total) return;
  int e = t & 15, lane = (t >> 4) & 31, nt = (t >> 9) & 7, kc = t >> 12;
  int k = kc * 32 + e + 8 * (lane >> 4) + ((e >> 3) << 3); // B layout: table order (e<8: e+8h; e>=8: 16+8h+e-8)
  int n = nt * 16 + (lane & 15);
  out[t] = f2bf(W[(size_t)k * 128 + n]);
}

// ---- grid embedding gather into concat buffer col 384..447 ----------------
__global__ void k_ge(const float* __restrict__ X, const float* __restrict__ ge,
                     unsigned short* __restrict__ cat){
  int t = blockIdx.x * 256 + threadIdx.x;       // B*64 threads
  int b = t >> 6, c = t & 63;
  int gidx = (int)(X[(size_t)b * 34 + 0] + X[(size_t)b * 34 + 1] * 100.0f);
  cat[(size_t)b * 448 + 384 + c] = f2bf(ge[(size_t)gidx * 64 + c]);
}

// ---- Xf = inp-MLP(X[:,2:34]) -> cat cols 0..127 ---------------------------
__global__ __launch_bounds__(256) void k_xf(
    const float* __restrict__ X, const unsigned short* __restrict__ W1,
    const float* __restrict__ b1, const unsigned short* __restrict__ W2,
    const float* __restrict__ b2, unsigned short* __restrict__ cat){
  __shared__ unsigned short hid[8][16 * 128];
  const int tid = threadIdx.x, w = tid >> 5, lane = tid & 31;
  const int l15 = lane & 15, half = lane >> 4;
  int row0 = (blockIdx.x * 8 + w) * 16;
  v8f acc1[8], acc2[8];
  Frag A; load_afrag_f32(X + (size_t)(row0 + l15) * 34 + 2, half, A);
  mlp_l1_fragA(A, W1, lane, acc1);
  hid_store(acc1, b1, hid[w], l15, half);
  mlp_l2(W2, hid[w], lane, l15, half, acc2);
#pragma unroll
  for (int nt = 0; nt < 8; ++nt){
    float bv = b2[nt * 16 + l15];
#pragma unroll
    for (int v = 0; v < 8; ++v)
      cat[(size_t)(row0 + v + 8 * half) * 448 + nt * 16 + l15] = f2bf(acc2[nt][v] + bv);
  }
}

// ---- qe = q-MLP(Xf) (f32 out, used for attention scores) ------------------
__global__ __launch_bounds__(256) void k_qe(
    const unsigned short* __restrict__ cat, const unsigned short* __restrict__ W1,
    const float* __restrict__ b1, const unsigned short* __restrict__ W2,
    const float* __restrict__ b2, float* __restrict__ qe){
  __shared__ unsigned short hid[8][16 * 128];
  const int tid = threadIdx.x, w = tid >> 5, lane = tid & 31;
  const int l15 = lane & 15, half = lane >> 4;
  int row0 = (blockIdx.x * 8 + w) * 16;
  v8f acc1[8], acc2[8];
  mlp_l1_bf16(cat + (size_t)(row0 + l15) * 448, W1, lane, half, acc1);
  hid_store(acc1, b1, hid[w], l15, half);
  mlp_l2(W2, hid[w], lane, l15, half, acc2);
#pragma unroll
  for (int nt = 0; nt < 8; ++nt){
    float bv = b2[nt * 16 + l15];
#pragma unroll
    for (int v = 0; v < 8; ++v)
      qe[(size_t)(row0 + v + 8 * half) * 128 + nt * 16 + l15] = acc2[nt][v] + bv;
  }
}

// ---- fused attention block: k-MLP -> scores -> softmax -> v-MLP -> output -
// 4 queries per workgroup; wave w owns neighbor tiles 2w, 2w+1 (16 nbrs each).
__global__ __launch_bounds__(256) void k_attn(
    const float* __restrict__ tf, const int* __restrict__ idx,
    const float* __restrict__ dist, const float* __restrict__ qe,
    const unsigned short* __restrict__ kW1, const float* __restrict__ kb1,
    const unsigned short* __restrict__ kW2,
    const unsigned short* __restrict__ vW1, const float* __restrict__ vb1,
    const unsigned short* __restrict__ vW2, const float* __restrict__ vb2,
    unsigned short* __restrict__ outcat, float thresh){
  __shared__ unsigned short hid[8][16 * 128];
  __shared__ float sco[4][8][64];
  __shared__ float osh[4][2][128];

  const int tid = threadIdx.x, w = tid >> 5, lane = tid & 31;
  const int l15 = lane & 15, half = lane >> 4;
  const int qbase = blockIdx.x * 4;
  v8f acc1[8], acc2[8];

  // Gather + convert this wave's two 16-neighbor A tiles ONCE; reuse in both
  // the k-MLP and v-MLP passes (halves gather traffic and bf16 pack VALU).
  Frag A1[2];
#pragma unroll
  for (int t = 0; t < 2; ++t){
    int rt = 2 * w + t, ql = rt >> 2, n0 = (rt & 3) * 16;
    int ridx = idx[(qbase + ql) * 64 + n0 + l15];
    load_afrag_f32(tf + (size_t)ridx * 32, half, A1[t]);
  }

  // pass 1: K-MLP + masked scores (k-MLP output bias is constant per (b,h)
  // across neighbors -> softmax-invariant -> dropped)
#pragma unroll 1
  for (int t = 0; t < 2; ++t){
    int rt = 2 * w + t, ql = rt >> 2, n0 = (rt & 3) * 16;
    int b = qbase + ql;
    mlp_l1_fragA(A1[t], kW1, lane, acc1);
    hid_store(acc1, kb1, hid[w], l15, half);
    mlp_l2(kW2, hid[w], lane, l15, half, acc2);
#pragma unroll
    for (int h = 0; h < 8; ++h){            // head h == output tile nt=h
      float qv = qe[(size_t)b * 128 + h * 16 + l15];
#pragma unroll
      for (int v = 0; v < 8; ++v){
        float s = qv * acc2[h][v];
        s += __shfl_xor(s, 1, 16);
        s += __shfl_xor(s, 2, 16);
        s += __shfl_xor(s, 4, 16);
        s += __shfl_xor(s, 8, 16);
        if (l15 == v){
          int n = n0 + v + 8 * half;
          sco[ql][h][n] = (dist[b * 64 + n] < thresh) ? (s * 0.25f) : -32767.0f;
        }
      }
    }
  }
  __syncthreads();
  if (tid < 32){                            // softmax per (query, head)
    int ql = tid >> 3, h = tid & 7;
    float m = -3.0e38f;
    for (int n = 0; n < 64; ++n) m = fmaxf(m, sco[ql][h][n]);
    float ssum = 0.f;
    for (int n = 0; n < 64; ++n){ float e = __expf(sco[ql][h][n] - m); sco[ql][h][n] = e; ssum += e; }
    float inv = 1.f / ssum;
    for (int n = 0; n < 64; ++n) sco[ql][h][n] *= inv;
  }
  __syncthreads();

  // pass 2: V-MLP, weight by softmax, deterministic 2-slot reduction
  float oreg[8];
#pragma unroll
  for (int h = 0; h < 8; ++h) oreg[h] = 0.f;
  int qlw = w >> 1;
#pragma unroll 1
  for (int t = 0; t < 2; ++t){
    int rt = 2 * w + t, ql = rt >> 2, n0 = (rt & 3) * 16;
    mlp_l1_fragA(A1[t], vW1, lane, acc1);
    hid_store(acc1, vb1, hid[w], l15, half);
    mlp_l2(vW2, hid[w], lane, l15, half, acc2);
#pragma unroll
    for (int h = 0; h < 8; ++h){
      const float* sp = &sco[ql][h][n0 + 8 * half];   // 8 contiguous weights
      float4 sa = *(const float4*)sp;
      float4 sb = *(const float4*)(sp + 4);
      float tsum = sa.x * acc2[h][0] + sa.y * acc2[h][1]
                 + sa.z * acc2[h][2] + sa.w * acc2[h][3]
                 + sb.x * acc2[h][4] + sb.y * acc2[h][5]
                 + sb.z * acc2[h][6] + sb.w * acc2[h][7];
      tsum += __shfl_xor(tsum, 16, 32);
      oreg[h] += tsum;
    }
  }
  if (half == 0){
#pragma unroll
    for (int h = 0; h < 8; ++h) osh[qlw][w & 1][h * 16 + l15] = oreg[h];
  }
  __syncthreads();
  {
    int ql = tid >> 6, f = (tid & 63) * 2;  // packed bf16 pair store
    float s0 = osh[ql][0][f]     + osh[ql][1][f]     + vb2[f];
    float s1 = osh[ql][0][f + 1] + osh[ql][1][f + 1] + vb2[f + 1];
    // sum includes v-MLP output bias once (sum(att)=1)
    *(unsigned int*)&outcat[(size_t)(qbase + ql) * 448 + f] = pack2(s0, s1);
  }
}

// ---- output head: Dense(448->128, elu) + Dense(128->1), scale -------------
__global__ __launch_bounds__(256) void k_out(
    const unsigned short* __restrict__ cat, const unsigned short* __restrict__ W1,
    const float* __restrict__ b1, const float* __restrict__ W2,
    const float* __restrict__ b2, float* __restrict__ out){
  const int tid = threadIdx.x, w = tid >> 5, lane = tid & 31;
  const int l15 = lane & 15, half = lane >> 4;
  int row0 = (blockIdx.x * 8 + w) * 16;
  const unsigned short* arow = cat + (size_t)(row0 + l15) * 448;
  v8f acc[8]; v8f z = {};
#pragma unroll
  for (int nt = 0; nt < 8; ++nt) acc[nt] = z;
#pragma unroll
  for (int kc = 0; kc < 14; ++kc){
    Frag A; load_afrag_bf16(arow + kc * 32, half, A);
#pragma unroll
    for (int nt = 0; nt < 8; ++nt){
      Frag B; load_bfrag(W1, kc, nt, lane, B);
      acc[nt] = wmma_bf(A, B, acc[nt]);
    }
  }
  float b1v[8], w2v[8];
#pragma unroll
  for (int nt = 0; nt < 8; ++nt){
    b1v[nt] = b1[nt * 16 + l15];
    w2v[nt] = W2[nt * 16 + l15];
  }
  float bb = b2[0];
#pragma unroll
  for (int v = 0; v < 8; ++v){
    float s = 0.f;
#pragma unroll
    for (int nt = 0; nt < 8; ++nt) s += eluf(acc[nt][v] + b1v[nt]) * w2v[nt];
    s += __shfl_xor(s, 1, 16);
    s += __shfl_xor(s, 2, 16);
    s += __shfl_xor(s, 4, 16);
    s += __shfl_xor(s, 8, 16);
    if (l15 == v) out[row0 + v + 8 * half] = (s + bb) * 0.8f + 12.0f;
  }
}

extern "C" void kernel_launch(void* const* d_in, const int* in_sizes, int n_in,
                              void* d_out, int out_size, void* d_ws, size_t ws_size,
                              hipStream_t stream){
  (void)in_sizes; (void)n_in; (void)out_size; (void)ws_size;
  const float* X     = (const float*)d_in[0];
  const int*   Nidx  = (const int*)d_in[1];
  const float* Ndist = (const float*)d_in[2];
  const int*   Eidx  = (const int*)d_in[3];
  const float* Edist = (const float*)d_in[4];
  const float* tf    = (const float*)d_in[5];
  const float* gemb  = (const float*)d_in[6];
  const float* P[32];
  for (int i = 0; i < 32; ++i) P[i] = (const float*)d_in[7 + i];
  // mlp slots in dict order: 0 inp, 1 q1, 2 k1, 3 v1, 4 q2, 5 k2, 6 v2, 7 out
#define WP1(m) P[(m)*4+0]
#define BP1(m) P[(m)*4+1]
#define WP2(m) P[(m)*4+2]
#define BP2(m) P[(m)*4+3]

  char* ws = (char*)d_ws; size_t off = 0;
  auto carve = [&](size_t bytes) -> void* {
    void* p = ws + off; off = (off + bytes + 255) & ~(size_t)255; return p;
  };

  static const int K1[8] = {32, 128, 32, 32, 128, 32, 32, 448};
  unsigned short* fw1[8]; unsigned short* fw2[8];
  for (int m = 0; m < 8; ++m){
    fw1[m] = (unsigned short*)carve((size_t)K1[m] * 128 * 2);
    fw2[m] = (m == 7) ? nullptr : (unsigned short*)carve((size_t)128 * 128 * 2);
  }
  unsigned short* cat = (unsigned short*)carve((size_t)4096 * 448 * 2); // [Xf|n1|n2|ge]
  float* qe1 = (float*)carve((size_t)4096 * 128 * 4);
  float* qe2 = (float*)carve((size_t)4096 * 128 * 4);

  for (int m = 0; m < 8; ++m){
    int tot1 = K1[m] * 128;
    k_prep<<<(tot1 + 255) / 256, 256, 0, stream>>>(WP1(m), fw1[m], tot1);
    if (m != 7){
      int tot2 = 128 * 128;
      k_prep<<<(tot2 + 255) / 256, 256, 0, stream>>>(WP2(m), fw2[m], tot2);
    }
  }
  k_ge<<<(4096 * 64) / 256, 256, 0, stream>>>(X, gemb, cat);
  k_xf<<<32, 256, 0, stream>>>(X, fw1[0], BP1(0), fw2[0], BP2(0), cat);
  k_qe<<<32, 256, 0, stream>>>(cat, fw1[1], BP1(1), fw2[1], BP2(1), qe1);
  k_qe<<<32, 256, 0, stream>>>(cat, fw1[4], BP1(4), fw2[4], BP2(4), qe2);
  k_attn<<<1024, 256, 0, stream>>>(tf, Nidx, Ndist, qe1, fw1[2], BP1(2), fw2[2],
                                   fw1[3], BP1(3), fw2[3], BP2(3), cat + 128, 0.5f);
  k_attn<<<1024, 256, 0, stream>>>(tf, Eidx, Edist, qe2, fw1[5], BP1(5), fw2[5],
                                   fw1[6], BP1(6), fw2[6], BP2(6), cat + 256, 0.02f);
  k_out<<<32, 256, 0, stream>>>(cat, fw1[7], BP1(7), WP2(7), BP2(7), (float*)d_out);
}
